// MultiHeadSelfAttention_36575941493022
// MI455X (gfx1250) — matrix operand order repaired and checked by reading, compiled, not run on hardware
//
#include <hip/hip_runtime.h>
#include <hip/hip_bf16.h>

#define BB 2
#define SS 2048
#define EE 2048
#define HH 16
#define DD 128
#define MM (BB * SS) // 4096

typedef __attribute__((ext_vector_type(16))) __bf16 v16bf;
typedef __attribute__((ext_vector_type(8))) float v8f;
typedef unsigned int v4u __attribute__((ext_vector_type(4)));
typedef unsigned int v8u __attribute__((ext_vector_type(8)));

union Frag16 {
  v16bf v;
  uint4 q[2];
  unsigned short u[16];
};

__device__ __forceinline__ unsigned short f2bf(float f) {
  unsigned u = __float_as_uint(f);
  u += 0x7FFFu + ((u >> 16) & 1u); // round-to-nearest-even
  return (unsigned short)(u >> 16);
}

__device__ __forceinline__ unsigned lds_addr32(const void* p) {
  // generic pointer to LDS: low 32 bits are the LDS byte address
  return (unsigned)(unsigned long long)(size_t)p;
}

__device__ __forceinline__ void async_load_b128(unsigned lds, const void* gaddr) {
  asm volatile("global_load_async_to_lds_b128 %0, %1, off" ::"v"(lds),
               "v"((unsigned long long)(size_t)gaddr)
               : "memory");
}

// ---------------- fp32 -> bf16 conversion ----------------
__global__ void cvt_f32_bf16(const float* __restrict__ in,
                             unsigned short* __restrict__ out, int n) {
  int i = (blockIdx.x * blockDim.x + threadIdx.x) * 4;
  if (i + 3 < n) {
    float4 f = *(const float4*)(in + i);
    unsigned lo = (unsigned)f2bf(f.x) | ((unsigned)f2bf(f.y) << 16);
    unsigned hi = (unsigned)f2bf(f.z) | ((unsigned)f2bf(f.w) << 16);
    *(uint2*)(out + i) = make_uint2(lo, hi);
  }
}

// ---------------- bf16 GEMM: C[M,N] = A[M,K] * W[N,K]^T + bias ----------------
// mode 0: write fp32 out[M,N]
// mode 1: write bf16 out in [B,H,S,D] layout (m = b*S+s, n = h*D+d)
__global__ __launch_bounds__(256) void gemm_bf16(
    const unsigned short* __restrict__ A, const unsigned short* __restrict__ W,
    const float* __restrict__ bias, void* __restrict__ out, int M, int N,
    int K, int mode) {
  __shared__ __align__(16) unsigned short As[128 * 32];
  __shared__ __align__(16) unsigned short Ws[128 * 32];

  const int tid = threadIdx.x;
  const int wave = tid >> 5, lane = tid & 31;
  const int lid = lane & 15, hi = lane >> 4;
  const int wm = wave >> 2, wn = wave & 3; // 2x4 wave grid, 64x32 per wave
  const int m0 = blockIdx.y * 128, n0 = blockIdx.x * 128;

  v8f acc[4][2] = {};

  const int srow = tid >> 1;       // 0..127
  const int scol = (tid & 1) * 16; // 0 or 16
  const unsigned la = lds_addr32(&As[srow * 32 + scol]);
  const unsigned lw = lds_addr32(&Ws[srow * 32 + scol]);

  for (int k0 = 0; k0 < K; k0 += 32) {
    __syncthreads(); // previous tile fully consumed before overwrite
    const unsigned short* ga = A + (size_t)(m0 + srow) * K + k0 + scol;
    const unsigned short* gw = W + (size_t)(n0 + srow) * K + k0 + scol;
    async_load_b128(la, ga);
    async_load_b128(la + 16, ga + 8);
    async_load_b128(lw, gw);
    async_load_b128(lw + 16, gw + 8);
    asm volatile("s_wait_asynccnt 0" ::: "memory");
    __syncthreads();

    Frag16 af[4], wf[2];
#pragma unroll
    for (int i = 0; i < 4; ++i) { // A-layout: lo lanes K{0-7,16-23}, hi K{8-15,24-31}
      int row = wm * 64 + i * 16 + lid;
      int c = hi * 8;
      af[i].q[0] = *(const uint4*)&As[row * 32 + c];
      af[i].q[1] = *(const uint4*)&As[row * 32 + c + 16];
    }
#pragma unroll
    for (int j = 0; j < 2; ++j) { // B-layout: lane=N col, lo lanes K{0-7,16-23}, hi K{8-15,24-31}
      int row = wn * 32 + j * 16 + lid;
      int c = hi * 8;
      wf[j].q[0] = *(const uint4*)&Ws[row * 32 + c];
      wf[j].q[1] = *(const uint4*)&Ws[row * 32 + c + 16];
    }
#pragma unroll
    for (int i = 0; i < 4; ++i)
#pragma unroll
      for (int j = 0; j < 2; ++j)
        acc[i][j] = __builtin_amdgcn_wmma_f32_16x16x32_bf16(
            false, af[i].v, false, wf[j].v, (short)0, acc[i][j], false, false);
  }

#pragma unroll
  for (int i = 0; i < 4; ++i)
#pragma unroll
    for (int j = 0; j < 2; ++j)
#pragma unroll
      for (int r = 0; r < 8; ++r) {
        int m = m0 + wm * 64 + i * 16 + r + hi * 8;
        int n = n0 + wn * 32 + j * 16 + lid;
        float val = acc[i][j][r] + bias[n];
        if (mode == 0) {
          ((float*)out)[(size_t)m * N + n] = val;
        } else {
          int b = m / SS, s = m % SS;
          int h = n >> 7, d = n & 127;
          ((unsigned short*)out)[(((size_t)(b * HH + h)) * SS + s) * DD + d] =
              f2bf(val);
        }
      }
}

// ---------------- causal flash attention (bf16 in, bf16 ctx out) ----------------
#define VTS 72 // padded k-stride of transposed V tile (rows stay 16B aligned)
__global__ __launch_bounds__(128) void attn_kernel(
    const unsigned short* __restrict__ Qb, const unsigned short* __restrict__ Kb,
    const unsigned short* __restrict__ Vb, unsigned short* __restrict__ ctxb) {
  __shared__ __align__(16) unsigned short Ks[64 * 128]; // [k][d]
  __shared__ __align__(16) unsigned short Vt[128 * VTS]; // [d][k] transposed
  __shared__ __align__(16) unsigned short Ps[4 * 16 * 64];

  const int tid = threadIdx.x;
  const int wave = tid >> 5, lane = tid & 31;
  const int lid = lane & 15, hi = lane >> 4;
  const int bh = blockIdx.y;   // 0..B*H-1
  const int qblk = blockIdx.x; // 0..S/64-1
  const int b = bh >> 4, h = bh & 15;
  const size_t base = (size_t)bh * SS * DD;
  const int q0 = qblk * 64 + wave * 16;
  const float rscale = 0.08838834764831845f; // 1/sqrt(D)

  // Q A-fragments, 4 chunks of K(=d)=32
  Frag16 qf[4];
  const unsigned short* qrow = Qb + base + (size_t)(q0 + lid) * DD;
#pragma unroll
  for (int c = 0; c < 4; ++c) {
    int off = c * 32 + hi * 8;
    qf[c].q[0] = *(const uint4*)(qrow + off);
    qf[c].q[1] = *(const uint4*)(qrow + off + 16);
  }

  v8f acc[8] = {};
  float mrow[8], lrow[8];
#pragma unroll
  for (int r = 0; r < 8; ++r) {
    mrow[r] = -3.0e38f;
    lrow[r] = 0.f;
  }

  const int nkb = qblk + 1; // causal: only blocks with k0 <= q
  const int crow = tid >> 1;       // k row 0..63
  const int ccol = (tid & 1) * 64; // d 0 or 64
  const unsigned ks_base = lds_addr32(&Ks[0]);

  for (int kb = 0; kb < nkb; ++kb) {
    const int k0 = kb * 64;
    __syncthreads();

    // --- K tile via Tensor Data Mover: one 2-group D# per 64x128 bf16 tile ---
    if (wave == 0) {
      unsigned long long ga =
          (unsigned long long)(size_t)(Kb + base + (size_t)k0 * DD);
      v4u g0;
      g0[0] = 1u;                   // count=1, is_restore=0, gather off
      g0[1] = ks_base;              // lds_addr
      g0[2] = (unsigned)ga;         // global_addr[31:0]
      g0[3] = (unsigned)(ga >> 32) | 0x80000000u; // addr[56:32] | type=2<<30
      v8u g1;
      g1[0] = 0x00010000u;          // wg_mask=0, data_size=1 (2B)
      g1[1] = 128u << 16;           // tensor_dim0 = 128 (low 16 in [31:16])
      g1[2] = 0xFFFF0000u;          // tensor_dim1 = 0xFFFF (no OOB clip)
      g1[3] = 128u << 16;           // tile_dim0 = 128
      g1[4] = 64u;                  // tile_dim1 = 64, tile_dim2 = 0
      g1[5] = 128u;                 // tensor_dim0_stride = 128 (48b lo)
      g1[6] = 0u;                   // stride hi, tensor_dim1_stride lo
      g1[7] = 0u;
      asm volatile("tensor_load_to_lds %0, %1" ::"s"(g0), "s"(g1) : "memory");
      asm volatile("s_wait_tensorcnt 0x0" ::: "memory");
    }

    // --- V tile: load coalesced, store transposed Vt[d][k] ---
    {
      const unsigned short* gv = Vb + base + (size_t)(k0 + crow) * DD + ccol;
#pragma unroll
      for (int qq = 0; qq < 8; ++qq) {
        union {
          uint4 q;
          unsigned short u[8];
        } t;
        t.q = *(const uint4*)(gv + qq * 8);
#pragma unroll
        for (int e = 0; e < 8; ++e)
          Vt[(ccol + qq * 8 + e) * VTS + crow] = t.u[e];
      }
    }
    __syncthreads();

    // S = Q * K^T : 4 k-tiles x 4 d-chunks
    v8f sacc[4] = {};
#pragma unroll
    for (int kt = 0; kt < 4; ++kt) {
#pragma unroll
      for (int dc = 0; dc < 4; ++dc) {
        Frag16 kf; // lane = k row (B col), d {0-7,16-23} lo / {8-15,24-31} hi from Ks row
        int addr = (kt * 16 + lid) * 128 + dc * 32 + hi * 8;
        kf.q[0] = *(const uint4*)&Ks[addr];
        kf.q[1] = *(const uint4*)&Ks[addr + 16];
        sacc[kt] = __builtin_amdgcn_wmma_f32_16x16x32_bf16(
            false, qf[dc].v, false, kf.v, (short)0, sacc[kt], false, false);
      }
    }

    // scale + causal mask + row max
    float rmax[8];
#pragma unroll
    for (int r = 0; r < 8; ++r) rmax[r] = -3.0e38f;
#pragma unroll
    for (int kt = 0; kt < 4; ++kt)
#pragma unroll
      for (int r = 0; r < 8; ++r) {
        int kg = k0 + kt * 16 + lid;
        int qg = q0 + r + hi * 8;
        float sv = (kg <= qg) ? sacc[kt][r] * rscale : -3.0e38f;
        sacc[kt][r] = sv;
        rmax[r] = fmaxf(rmax[r], sv);
      }
#pragma unroll
    for (int r = 0; r < 8; ++r) {
#pragma unroll
      for (int msk = 1; msk < 16; msk <<= 1)
        rmax[r] = fmaxf(rmax[r], __shfl_xor(rmax[r], msk, 32));
      float mnew = fmaxf(mrow[r], rmax[r]);
      float alpha = __expf(mrow[r] - mnew);
      mrow[r] = mnew;
      lrow[r] *= alpha;
#pragma unroll
      for (int dt = 0; dt < 8; ++dt) acc[dt][r] *= alpha;
    }

    // P = exp(S - m), stash bf16 to per-wave LDS (C-layout -> memory)
    float rsum[8];
#pragma unroll
    for (int r = 0; r < 8; ++r) rsum[r] = 0.f;
#pragma unroll
    for (int kt = 0; kt < 4; ++kt)
#pragma unroll
      for (int r = 0; r < 8; ++r) {
        float p = __expf(sacc[kt][r] - mrow[r]);
        rsum[r] += p;
        Ps[wave * 1024 + (r + hi * 8) * 64 + kt * 16 + lid] = f2bf(p);
      }
#pragma unroll
    for (int r = 0; r < 8; ++r) {
#pragma unroll
      for (int msk = 1; msk < 16; msk <<= 1)
        rsum[r] += __shfl_xor(rsum[r], msk, 32);
      lrow[r] += rsum[r];
    }

    asm volatile("s_wait_dscnt 0" ::: "memory"); // wave-local P write->read

    Frag16 pf[2]; // reload P in A-layout
#pragma unroll
    for (int c = 0; c < 2; ++c) {
      int off = wave * 1024 + lid * 64 + c * 32 + hi * 8;
      pf[c].q[0] = *(const uint4*)&Ps[off];
      pf[c].q[1] = *(const uint4*)&Ps[off + 16];
    }

    // ctx += P * V  (V B-fragments now two ds_load_b128 from transposed tile)
#pragma unroll
    for (int dt = 0; dt < 8; ++dt) {
#pragma unroll
      for (int c = 0; c < 2; ++c) {
        Frag16 vf; // lane = d col = Vt row; k {0-7,16-23} lo / {8-15,24-31} hi
        int addr = (dt * 16 + lid) * VTS + c * 32 + hi * 8;
        vf.q[0] = *(const uint4*)&Vt[addr];
        vf.q[1] = *(const uint4*)&Vt[addr + 16];
        acc[dt] = __builtin_amdgcn_wmma_f32_16x16x32_bf16(
            false, pf[c].v, false, vf.v, (short)0, acc[dt], false, false);
      }
    }
  }

  // normalize, merge heads -> bf16 ctx [B,S,E]
#pragma unroll
  for (int r = 0; r < 8; ++r) {
    float inv = 1.0f / lrow[r];
    int s = q0 + r + hi * 8;
#pragma unroll
    for (int dt = 0; dt < 8; ++dt) {
      int col = h * DD + dt * 16 + lid;
      ctxb[((size_t)(b * SS + s)) * EE + col] = f2bf(acc[dt][r] * inv);
    }
  }
}

extern "C" void kernel_launch(void* const* d_in, const int* in_sizes, int n_in,
                              void* d_out, int out_size, void* d_ws,
                              size_t ws_size, hipStream_t stream) {
  (void)in_sizes; (void)n_in; (void)out_size; (void)ws_size;
  const float* x  = (const float*)d_in[0];
  const float* Wq = (const float*)d_in[1];
  const float* bq = (const float*)d_in[2];
  const float* Wk = (const float*)d_in[3];
  const float* bk = (const float*)d_in[4];
  const float* Wv = (const float*)d_in[5];
  const float* bv = (const float*)d_in[6];
  const float* Wo = (const float*)d_in[7];
  const float* bo = (const float*)d_in[8];
  float* out = (float*)d_out;

  char* ws = (char*)d_ws;
  size_t off = 0;
  auto alloc = [&](size_t bytes) {
    char* p = ws + off;
    off += (bytes + 255) & ~(size_t)255;
    return p;
  };
  unsigned short* xb   = (unsigned short*)alloc((size_t)MM * EE * 2);
  unsigned short* wqb  = (unsigned short*)alloc((size_t)EE * EE * 2);
  unsigned short* wkb  = (unsigned short*)alloc((size_t)EE * EE * 2);
  unsigned short* wvb  = (unsigned short*)alloc((size_t)EE * EE * 2);
  unsigned short* wob  = (unsigned short*)alloc((size_t)EE * EE * 2);
  unsigned short* Qb   = (unsigned short*)alloc((size_t)MM * EE * 2);
  unsigned short* Kb   = (unsigned short*)alloc((size_t)MM * EE * 2);
  unsigned short* Vb   = (unsigned short*)alloc((size_t)MM * EE * 2);
  unsigned short* ctxb = (unsigned short*)alloc((size_t)MM * EE * 2);

  const int thr = 256;
  int nx = MM * EE;
  int nw = EE * EE;
  cvt_f32_bf16<<<nx / (thr * 4), thr, 0, stream>>>(x, xb, nx);
  cvt_f32_bf16<<<nw / (thr * 4), thr, 0, stream>>>(Wq, wqb, nw);
  cvt_f32_bf16<<<nw / (thr * 4), thr, 0, stream>>>(Wk, wkb, nw);
  cvt_f32_bf16<<<nw / (thr * 4), thr, 0, stream>>>(Wv, wvb, nw);
  cvt_f32_bf16<<<nw / (thr * 4), thr, 0, stream>>>(Wo, wob, nw);

  dim3 gg(EE / 128, MM / 128);
  gemm_bf16<<<gg, 256, 0, stream>>>(xb, wqb, bq, Qb, MM, EE, EE, 1);
  gemm_bf16<<<gg, 256, 0, stream>>>(xb, wkb, bk, Kb, MM, EE, EE, 1);
  gemm_bf16<<<gg, 256, 0, stream>>>(xb, wvb, bv, Vb, MM, EE, EE, 1);

  attn_kernel<<<dim3(SS / 64, BB * HH), 128, 0, stream>>>(Qb, Kb, Vb, ctxb);

  gemm_bf16<<<gg, 256, 0, stream>>>(ctxb, wob, bo, out, MM, EE, EE, 0);
}
